// GraphLineupModel_15547781611844
// MI455X (gfx1250) — matrix operand order repaired and checked by reading, not compiled, not run on hardware
//
#include <hip/hip_runtime.h>
#include <hip/hip_bf16.h>
#include <math.h>
#include <stdint.h>

// ---------------------------------------------------------------------------
// GraphLineupModel for MI455X (gfx1250, wave32). One workgroup per graph.
// GEMMs: v_wmma_f32_16x16x32_f16, f16 operands in LDS in fragment-native
// layouts (A row-major: 2x ds_load_b128/frag; B K-major: 2x ds_load_b128/frag,
// shared across the wave's 4 row-tiles). C tiles live in VGPRs across K-chunks.
// Per-graph inputs staged via global_load_async_to_lds_b128 (ASYNCcnt);
// next-stage weights warmed with global_prefetch_b8 while WMMAs run.
// ---------------------------------------------------------------------------

#define NN        64
#define F_INN     32
#define DD        128
#define PHI_D     256
#define NWAVES    8
#define NTHREADS  256
#define LBLK      3

typedef __attribute__((ext_vector_type(16))) _Float16 v16h;
typedef __attribute__((ext_vector_type(8)))  _Float16 h8;   // 16 bytes
typedef __attribute__((ext_vector_type(8)))  float    v8f;

struct KParams {
  const float* adj;   // [B,64,64]
  const float* feat;  // [B,64,32]
  const float* home;  // [B,64]
  const float* blk_b1[LBLK];
  const float* blk_b2[LBLK];
  const float* blk_rgcn_b[LBLK];
  const float* blk_root[LBLK];
  const float* blk_rgcn_w[LBLK];  // [2,128,128]: rel0=neg, rel1=pos
  const float* blk_w1[LBLK];
  const float* blk_w2[LBLK];
  const float* embed_b; const float* embed_w;   // [32,128]
  const float* ln_b;    const float* ln_g;
  const float* phi_b1;  const float* phi_b2;
  const float* phi_w1;  const float* phi_w2;    // [128,256],[256,256]
  const float* proj_b;  const float* proj_w;    // [128,128]
  const float* rho_b1;  const float* rho_b2;
  const float* rho_w1;  const float* rho_w2;    // [256,128],[128,1]
  float* out;                                    // [B]
};

// --- CDNA5 async copy: global -> LDS, tracked by ASYNCcnt -------------------
__device__ __forceinline__ void async_g2l_b128(void* lds, const void* gaddr) {
  const uint32_t ldsOff = (uint32_t)(uintptr_t)lds;   // flat addr[31:0] = LDS offset
  asm volatile("global_load_async_to_lds_b128 %0, %1, off"
               :: "v"(ldsOff), "v"(gaddr) : "memory");
}
__device__ __forceinline__ void wait_async0() {
  asm volatile("s_wait_asynccnt 0x0" ::: "memory");
}

// warm L2 for an upcoming weight matrix (one prefetch per 128B line)
__device__ __forceinline__ void prefetch_global(const void* p, int bytes, int tid) {
  const char* c = (const char*)p;
  for (int off = tid * 128; off < bytes; off += NTHREADS * 128)
    __builtin_prefetch(c + off, 0, 0);
}

// --- fragment loaders: packed 16B LDS loads, zero converts ------------------
__device__ __forceinline__ v16h load_a16(const _Float16* A, int lda,
                                         int m, int k0, int lane) {
  const int kb = (lane & 16) ? 8 : 0;
  const _Float16* p = A + m * lda + k0 + kb;
  h8 lo = *(const h8*)p;
  h8 hi = *(const h8*)(p + 16);
  v16h a;
#pragma unroll
  for (int e = 0; e < 8; ++e) { a[e] = lo[e]; a[e + 8] = hi[e]; }
  return a;
}

__device__ __forceinline__ v16h load_bt16(const _Float16* Bt, int ldk,
                                          int n, int k0, int lane) {
  const int kb = (lane & 16) ? 8 : 0;
  const _Float16* p = Bt + n * ldk + k0 + kb;
  h8 b0 = *(const h8*)p;
  h8 b1 = *(const h8*)(p + 16);
  v16h b;
#pragma unroll
  for (int e = 0; e < 8; ++e) { b[e] = b0[e]; b[e + 8] = b1[e]; }
  return b;
}

template <int TPW>
__device__ __forceinline__ void init_tiles(v8f (&acc)[TPW], const float* bias,
                                           int Nt, int wave, int lane) {
#pragma unroll
  for (int i = 0; i < TPW; ++i) {
    const int nt = (wave + i * NWAVES) % Nt;
    const float bv = bias ? bias[nt * 16 + (lane & 15)] : 0.0f;
#pragma unroll
    for (int r = 0; r < 8; ++r) acc[i][r] = bv;
  }
}

// acc += A[M,K] @ Bt^T ; tiles t = wave + i*NWAVES (fixed column per wave for
// Nt==NWAVES -> B fragment hoisted and reused across the TPW row tiles)
template <int TPW>
__device__ __forceinline__ void gemm_acc(v8f (&acc)[TPW],
                                         const _Float16* A, int lda,
                                         const _Float16* Bt, int ldk,
                                         int K, int Nt, int wave, int lane) {
#pragma unroll
  for (int i = 0; i < TPW; ++i) {
    const int t = wave + i * NWAVES;
    const int mt = t / Nt, nt = t % Nt;
    const int m = mt * 16 + (lane & 15);
    const int n = nt * 16 + (lane & 15);
    for (int k0 = 0; k0 < K; k0 += 32) {
      v16h a = load_a16(A, lda, m, k0, lane);
      v16h b = load_bt16(Bt, ldk, n, k0, lane);
      acc[i] = __builtin_amdgcn_wmma_f32_16x16x32_f16(false, a, false, b,
                                                      (short)0, acc[i], false, false);
    }
  }
}

// store C tiles as f16; layout selection folded at compile time.
// transposed: lane holds fixed n, m contiguous -> one 16B ds_store_b128/tile.
template <int TPW, bool RM, bool TR, bool RELU>
__device__ __forceinline__ void store_tiles(const v8f (&acc)[TPW],
                                            _Float16* Crm, int ldc,
                                            _Float16* Ct, int ldm,
                                            int Nt, int wave, int lane) {
#pragma unroll
  for (int i = 0; i < TPW; ++i) {
    const int t = wave + i * NWAVES;
    const int mt = t / Nt, nt = t % Nt;
    const int n  = nt * 16 + (lane & 15);
    const int mb = mt * 16 + ((lane & 16) ? 8 : 0);
    h8 o;
#pragma unroll
    for (int r = 0; r < 8; ++r) {
      float v = acc[i][r];
      if (RELU) v = fmaxf(v, 0.0f);
      o[r] = (_Float16)v;
    }
    if (TR) *(h8*)(Ct + n * ldm + mb) = o;
    if (RM) {
#pragma unroll
      for (int r = 0; r < 8; ++r) Crm[(size_t)(mb + r) * ldc + n] = o[r];
    }
  }
}

// stage global f32 weight W[K][N] columns [n0,n0+Nsub) as f16 transposed
__device__ __forceinline__ void stage_wt(const float* __restrict__ W, int N,
                                         int Ksub, int n0, int Nsub,
                                         _Float16* __restrict__ Wt, int tid) {
  for (int x = tid; x < Nsub * Ksub; x += NTHREADS) {
    const int k = x / Nsub, nl = x % Nsub;
    Wt[nl * Ksub + k] = (_Float16)W[(size_t)k * N + n0 + nl];
  }
}

__global__ __launch_bounds__(NTHREADS, 1)
void graph_lineup_kernel(KParams kp) {
  extern __shared__ char smemraw[];
  _Float16* sPneg = (_Float16*)smemraw;   // [64][64]  P_neg[j][i]=1[a<0]/deg
  _Float16* sPpos = sPneg + 4096;         // [64][64]
  _Float16* sAabs = sPpos + 4096;         // [64][64]  |a|[i][j]
  _Float16* sHidR = sAabs + 4096;         // [64][128] hidden row-major
  _Float16* sHidT = sHidR + 8192;         // [128][64] hidden K-major
  _Float16* sT1   = sHidT + 8192;         // [64][128] Tneg / T1 / ph1 lo half
  _Float16* sT2   = sT1 + 8192;           // [64][128] Tpos / ph1 hi half
  _Float16* sMT   = sT2 + 8192;           // [128][64] M^T
  _Float16* sAgg  = sMT + 8192;           // [64][128] aggregated
  _Float16* sWt   = sAgg + 8192;          // 16384 f16 = 32KB weight staging
  float*    sRed  = (float*)(sWt + 16384);// 768 f32: sums / stats / reduce
  _Float16* sPh1  = sT1;                  // [64][256] overlays T1+T2

  const int b    = blockIdx.x;
  const int tid  = threadIdx.x;
  const int lane = tid & 31;
  const int wave = tid >> 5;

  // ---- async-stage per-graph inputs (adj 16KB + feat 8KB) into LDS -------
  float* fadj  = (float*)sHidR;           // 4096 f32 (sHidR region)
  float* ffeat = fadj + NN * NN;          // 2048 f32 (sHidT region)
  {
    const float* ga = kp.adj  + (size_t)b * NN * NN;
    const float* gf = kp.feat + (size_t)b * NN * F_INN;
    for (int x = tid * 4; x < NN * NN; x += NTHREADS * 4)
      async_g2l_b128(fadj + x, ga + x);
    for (int x = tid * 4; x < NN * F_INN; x += NTHREADS * 4)
      async_g2l_b128(ffeat + x, gf + x);
    prefetch_global(kp.embed_w, F_INN * DD * 4, tid);
    wait_async0();
  }
  __syncthreads();

  // ---- adjacency -> P_neg/P_pos (deg-folded, transposed), |A| ------------
  if (tid < NN) {                         // column sums of edge indicators
    float cp = 0.f, cn = 0.f;
    for (int i = 0; i < NN; ++i) {
      const float v = fadj[i * NN + tid];
      cp += (v > 0.f) ? 1.f : 0.f;
      cn += (v < 0.f) ? 1.f : 0.f;
    }
    sRed[512 + tid] = 1.0f / fmaxf(cp, 1.0f);
    sRed[576 + tid] = 1.0f / fmaxf(cn, 1.0f);
  }
  __syncthreads();
  for (int x = tid; x < NN * NN; x += NTHREADS) {
    const int i = x >> 6, j = x & 63;
    const float v = fadj[x];
    sPpos[j * NN + i] = (_Float16)((v > 0.f) ? sRed[512 + j] : 0.f);
    sPneg[j * NN + i] = (_Float16)((v < 0.f) ? sRed[576 + j] : 0.f);
    sAabs[x] = (_Float16)fabsf(v);
  }
  for (int x = tid; x < NN * F_INN; x += NTHREADS)
    sAgg[x] = (_Float16)ffeat[x];                       // feat f16 row-major
  stage_wt(kp.embed_w, DD, F_INN, 0, DD, sWt, tid);     // embed_w^T
  prefetch_global(kp.proj_w, DD * DD * 4, tid);
  __syncthreads();

  // ---- embed + proj ------------------------------------------------------
  {
    v8f acc[4];
    init_tiles<4>(acc, kp.embed_b, 8, wave, lane);
    gemm_acc<4>(acc, sAgg, F_INN, sWt, F_INN, F_INN, 8, wave, lane);
    store_tiles<4, true, false, true>(acc, sT1, DD, nullptr, 0, 8, wave, lane);
  }
  __syncthreads();
  stage_wt(kp.proj_w, DD, DD, 0, DD, sWt, tid);
  prefetch_global(kp.blk_rgcn_w[0], DD * DD * 4, tid);
  __syncthreads();
  {
    v8f acc[4];
    init_tiles<4>(acc, kp.proj_b, 8, wave, lane);
    gemm_acc<4>(acc, sT1, DD, sWt, DD, DD, 8, wave, lane);
    store_tiles<4, true, true, false>(acc, sHidR, DD, sHidT, NN, 8, wave, lane);
  }
  for (int x = tid; x < NN * DD; x += NTHREADS) sAgg[x] = (_Float16)0.0f;
  __syncthreads();

  // ---- GCN blocks --------------------------------------------------------
  for (int l = 0; l < LBLK; ++l) {
    // hidden += aggregated (maintain both hidden layouts)
    for (int x = tid; x < NN * DD; x += NTHREADS) {
      const float h = (float)sHidR[x] + (float)sAgg[x];
      const _Float16 hh = (_Float16)h;
      sHidR[x] = hh;
      sHidT[(x & 127) * NN + (x >> 7)] = hh;
    }
    __syncthreads();
    // Tneg = P_neg @ hidden ; Tpos = P_pos @ hidden (disjoint outputs)
    {
      v8f acc[4];
      init_tiles<4>(acc, nullptr, 8, wave, lane);
      gemm_acc<4>(acc, sPneg, NN, sHidT, NN, NN, 8, wave, lane);
      store_tiles<4, true, false, false>(acc, sT1, DD, nullptr, 0, 8, wave, lane);
    }
    {
      v8f acc[4];
      init_tiles<4>(acc, nullptr, 8, wave, lane);
      gemm_acc<4>(acc, sPpos, NN, sHidT, NN, NN, 8, wave, lane);
      store_tiles<4, true, false, false>(acc, sT2, DD, nullptr, 0, 8, wave, lane);
    }
    __syncthreads();
    // M = Tneg@W0 + Tpos@W1 + hidden@root + b  (C resident in VGPRs)
    {
      v8f macc[4];
      init_tiles<4>(macc, kp.blk_rgcn_b[l], 8, wave, lane);
      stage_wt(kp.blk_rgcn_w[l], DD, DD, 0, DD, sWt, tid);
      prefetch_global(kp.blk_rgcn_w[l] + DD * DD, DD * DD * 4, tid);
      __syncthreads();
      gemm_acc<4>(macc, sT1, DD, sWt, DD, DD, 8, wave, lane);
      __syncthreads();
      stage_wt(kp.blk_rgcn_w[l] + DD * DD, DD, DD, 0, DD, sWt, tid);
      prefetch_global(kp.blk_root[l], DD * DD * 4, tid);
      __syncthreads();
      gemm_acc<4>(macc, sT2, DD, sWt, DD, DD, 8, wave, lane);
      __syncthreads();
      stage_wt(kp.blk_root[l], DD, DD, 0, DD, sWt, tid);
      prefetch_global(kp.blk_w1[l], DD * DD * 4, tid);
      __syncthreads();
      gemm_acc<4>(macc, sHidR, DD, sWt, DD, DD, 8, wave, lane);
      store_tiles<4, false, true, false>(macc, nullptr, 0, sMT, NN, 8, wave, lane);
    }
    __syncthreads();
    // aggregated = |A| @ M
    {
      v8f acc[4];
      init_tiles<4>(acc, nullptr, 8, wave, lane);
      gemm_acc<4>(acc, sAabs, NN, sMT, NN, NN, 8, wave, lane);
      store_tiles<4, true, false, false>(acc, sAgg, DD, nullptr, 0, 8, wave, lane);
    }
    __syncthreads();
    // LayerNorm + relu (row-wise over D=128, f32 math)
    if (tid < NN) {
      const _Float16* row = sAgg + tid * DD;
      float m = 0.f;
      for (int c = 0; c < DD; ++c) m += (float)row[c];
      m *= (1.0f / DD);
      float v = 0.f;
      for (int c = 0; c < DD; ++c) { const float d = (float)row[c] - m; v += d * d; }
      v *= (1.0f / DD);
      sRed[512 + tid] = m;
      sRed[576 + tid] = rsqrtf(v + 1e-5f);
    }
    __syncthreads();
    for (int x = tid; x < NN * DD; x += NTHREADS) {
      const int r = x >> 7, c = x & 127;
      const float v = ((float)sAgg[x] - sRed[512 + r]) * sRed[576 + r]
                      * kp.ln_g[c] + kp.ln_b[c];
      sAgg[x] = (_Float16)fmaxf(v, 0.0f);
    }
    stage_wt(kp.blk_w1[l], DD, DD, 0, DD, sWt, tid);
    prefetch_global(kp.blk_w2[l], DD * DD * 4, tid);
    __syncthreads();
    // FF: T1 = relu(agg @ w1 + b1) ; agg = T1 @ w2 + b2
    {
      v8f acc[4];
      init_tiles<4>(acc, kp.blk_b1[l], 8, wave, lane);
      gemm_acc<4>(acc, sAgg, DD, sWt, DD, DD, 8, wave, lane);
      store_tiles<4, true, false, true>(acc, sT1, DD, nullptr, 0, 8, wave, lane);
    }
    __syncthreads();
    stage_wt(kp.blk_w2[l], DD, DD, 0, DD, sWt, tid);
    prefetch_global((l + 1 < LBLK) ? kp.blk_rgcn_w[l + 1] : kp.phi_w1,
                    DD * DD * 4, tid);
    __syncthreads();
    {
      v8f acc[4];
      init_tiles<4>(acc, kp.blk_b2[l], 8, wave, lane);
      gemm_acc<4>(acc, sT1, DD, sWt, DD, DD, 8, wave, lane);
      store_tiles<4, true, false, false>(acc, sAgg, DD, nullptr, 0, 8, wave, lane);
    }
    __syncthreads();
  }
  for (int x = tid; x < NN * DD; x += NTHREADS)
    sHidR[x] = (_Float16)((float)sHidR[x] + (float)sAgg[x]);
  __syncthreads();

  // ---- phi1: ph1 = relu(hidden @ phi_w1 + b1)  [64x256], two N-halves ----
  for (int half = 0; half < 2; ++half) {
    stage_wt(kp.phi_w1, PHI_D, DD, half * DD, DD, sWt, tid);
    prefetch_global(kp.phi_w2 + (size_t)half * DD * PHI_D, DD * PHI_D * 4, tid);
    __syncthreads();
    v8f acc[4];
    init_tiles<4>(acc, kp.phi_b1 + half * DD, 8, wave, lane);
    gemm_acc<4>(acc, sHidR, DD, sWt, DD, DD, 8, wave, lane);
    store_tiles<4, true, false, true>(acc, sPh1 + half * DD, PHI_D, nullptr, 0,
                                      8, wave, lane);
    __syncthreads();
  }
  for (int x = tid; x < 2 * PHI_D; x += NTHREADS) sRed[x] = 0.0f;
  __syncthreads();

  // ---- fused ph2 GEMM + masked set-sums (ds_add_f32) ---------------------
  {
    float* sHome = sRed;
    float* sAway = sRed + PHI_D;
    for (int nhalf = 0; nhalf < 2; ++nhalf) {
      v8f acc[4];
      init_tiles<4>(acc, kp.phi_b2 + nhalf * DD, 8, wave, lane);
      for (int khalf = 0; khalf < 2; ++khalf) {       // C stays in VGPRs
        stage_wt(kp.phi_w2 + (size_t)khalf * DD * PHI_D, PHI_D, DD,
                 nhalf * DD, DD, sWt, tid);
        __syncthreads();
        gemm_acc<4>(acc, sPh1 + khalf * DD, PHI_D, sWt, DD, DD, 8, wave, lane);
        __syncthreads();
      }
#pragma unroll
      for (int i = 0; i < 4; ++i) {
        const int t  = wave + i * NWAVES;
        const int mt = t / 8, nt = t % 8;
        const int n  = nhalf * DD + nt * 16 + (lane & 15);
        const int mb = mt * 16 + ((lane & 16) ? 8 : 0);
        float hacc = 0.f, aacc = 0.f;
#pragma unroll
        for (int r = 0; r < 8; ++r) {
          const float v  = fmaxf(acc[i][r], 0.0f);
          const float hm = kp.home[(size_t)b * NN + mb + r];
          hacc += v * hm;
          aacc += v * (1.0f - hm);
        }
        atomicAdd(&sHome[n], hacc);
        atomicAdd(&sAway[n], aacc);
      }
    }
  }
  __syncthreads();

  // ---- rho head: tid<128 -> home, tid>=128 -> away -----------------------
  {
    const int t = tid & 127;
    const float* x = (tid < 128) ? sRed : (sRed + PHI_D);
    float acc = kp.rho_b1[t];
    for (int k = 0; k < PHI_D; ++k) acc += x[k] * kp.rho_w1[k * 128 + t];
    acc = fmaxf(acc, 0.0f);
    sRed[512 + tid] = acc * kp.rho_w2[t];
  }
  __syncthreads();
  for (int s = 64; s > 0; s >>= 1) {
    if ((tid & 127) < s) sRed[512 + tid] += sRed[512 + tid + s];
    __syncthreads();
  }
  if (tid == 0) {
    const float score = sRed[512] - sRed[512 + 128];   // rho_b2 cancels
    kp.out[b] = 0.5f + 0.5f * tanhf(score);
  }
}

extern "C" void kernel_launch(void* const* d_in, const int* in_sizes, int n_in,
                              void* d_out, int out_size, void* d_ws, size_t ws_size,
                              hipStream_t stream) {
  (void)in_sizes; (void)n_in; (void)d_ws; (void)ws_size; (void)out_size;
  KParams kp;
  // jax pytree order (dict keys sorted): adjacency, features, home_mask, params{
  //   blocks[0..2]{b1,b2,rgcn_b,rgcn_root,rgcn_w,w1,w2}, embed_b, embed_w,
  //   ln_b, ln_g, phi_b1, phi_b2, phi_w1, phi_w2, proj_b, proj_w,
  //   rho_b1, rho_b2, rho_w1, rho_w2 }
  kp.adj  = (const float*)d_in[0];
  kp.feat = (const float*)d_in[1];
  kp.home = (const float*)d_in[2];
  int i = 3;
  for (int l = 0; l < LBLK; ++l) {
    kp.blk_b1[l]     = (const float*)d_in[i + 0];
    kp.blk_b2[l]     = (const float*)d_in[i + 1];
    kp.blk_rgcn_b[l] = (const float*)d_in[i + 2];
    kp.blk_root[l]   = (const float*)d_in[i + 3];
    kp.blk_rgcn_w[l] = (const float*)d_in[i + 4];
    kp.blk_w1[l]     = (const float*)d_in[i + 5];
    kp.blk_w2[l]     = (const float*)d_in[i + 6];
    i += 7;
  }
  kp.embed_b = (const float*)d_in[i + 0];
  kp.embed_w = (const float*)d_in[i + 1];
  kp.ln_b    = (const float*)d_in[i + 2];
  kp.ln_g    = (const float*)d_in[i + 3];
  kp.phi_b1  = (const float*)d_in[i + 4];
  kp.phi_b2  = (const float*)d_in[i + 5];
  kp.phi_w1  = (const float*)d_in[i + 6];
  kp.phi_w2  = (const float*)d_in[i + 7];
  kp.proj_b  = (const float*)d_in[i + 8];
  kp.proj_w  = (const float*)d_in[i + 9];
  kp.rho_b1  = (const float*)d_in[i + 10];
  kp.rho_b2  = (const float*)d_in[i + 11];
  kp.rho_w1  = (const float*)d_in[i + 12];
  kp.rho_w2  = (const float*)d_in[i + 13];
  kp.out = (float*)d_out;

  // LDS: 3*4096 + 6*8192 + 16384 f16  +  768 f32  = 158,720 bytes
  const size_t smemBytes = (size_t)(3 * 4096 + 6 * 8192 + 16384) * sizeof(_Float16)
                         + 768 * sizeof(float);
  (void)hipFuncSetAttribute((const void*)graph_lineup_kernel,
                            hipFuncAttributeMaxDynamicSharedMemorySize,
                            (int)smemBytes);
  hipLaunchKernelGGL(graph_lineup_kernel, dim3(2048), dim3(NTHREADS),
                     smemBytes, stream, kp);
}
